// BinarySEResBlock_30588757082563
// MI455X (gfx1250) — matrix operand order repaired and checked by reading, not compiled, not run on hardware
//
#include <hip/hip_runtime.h>
#include <hip/hip_fp16.h>

typedef __attribute__((ext_vector_type(16))) _Float16 v16h;
typedef __attribute__((ext_vector_type(8)))  _Float16 v8h;
typedef __attribute__((ext_vector_type(8)))  float    v8f;

#define BATCH 32
#define CCH   256
#define LLEN  4096
#define NLTOT (BATCH*LLEN)     // per-channel element count for BN stats
#define KP    40               // padded LDS row stride in halves (80B rows)
#define TM    128
#define TN    128
#define KC    32               // WMMA K per step

union AF { v16h v; v8h h[2]; };

// ---------------- zero stats scratch ----------------
__global__ void k_zero(float* __restrict__ p, int n) {
  int i = blockIdx.x * blockDim.x + threadIdx.x;
  if (i < n) p[i] = 0.f;
}

// ---------------- binarize weights: [C][Cin][3] -> sign as f16 [k][C][Cin] ----------------
__global__ void k_sign(const float* __restrict__ w1, const float* __restrict__ w2,
                       _Float16* __restrict__ s1, _Float16* __restrict__ s2) {
  int i = blockIdx.x * blockDim.x + threadIdx.x;
  if (i >= 3 * CCH * CCH) return;
  int ci = i & (CCH - 1);
  int co = (i >> 8) & (CCH - 1);
  int k  = i >> 16;
  int src = (co * CCH + ci) * 3 + k;
  float a = w1[src], b = w2[src];
  s1[i] = (_Float16)((a > 0.f) ? 1.f : ((a < 0.f) ? -1.f : 0.f));
  s2[i] = (_Float16)((b > 0.f) ? 1.f : ((b < 0.f) ? -1.f : 0.f));
}

// ---------------- implicit-GEMM conv via WMMA, double-buffered LDS ----------------
// MODE 0: conv1 (raw fp32 input), accumulate chan sum/sumsq
// MODE 1: conv2 (input = clamp(scale1*y1+shift1)), also per-(b,ch) row sums for SE
template<int MODE>
__global__ __launch_bounds__(256) void k_conv_gemm(
    const float* __restrict__ src, const _Float16* __restrict__ wsgn,
    float* __restrict__ dst,
    const float* __restrict__ sc1, const float* __restrict__ sh1,
    float* __restrict__ csum, float* __restrict__ csumsq, float* __restrict__ rsum)
{
  __shared__ __align__(32) _Float16 Bs[2][130 * KP];

  const int tid  = threadIdx.x;
  const int lane = tid & 31;
  const int wave = tid >> 5;
  const int wm   = wave & 3;      // 4 M-groups of 32 channels
  const int wn   = wave >> 2;     // 2 N-groups of 64 positions
  const int m0   = blockIdx.x * TM;
  const int l0   = blockIdx.y * TN;
  const int b    = blockIdx.z;
  const int col  = lane & 15;
  const int hi   = lane >> 4;     // 0: lanes 0-15, 1: lanes 16-31

  v8f zero = {};
  v8f acc[2][4];
  #pragma unroll
  for (int mf = 0; mf < 2; ++mf)
    #pragma unroll
    for (int nf = 0; nf < 4; ++nf) acc[mf][nf] = zero;

  const float* srcB = src + (size_t)b * CCH * LLEN;

  // tile loader: thread t owns channel c = t>>3, loads float4s along l.
  // float4 index f in [0,34) covers gl = l0-4+4f .. +3 ; LDS row ll = 4f+j-3.
  const int lc = tid >> 3;        // channel within chunk (0..31)
  const int lq = tid & 7;         // float4 phase (0..7)
  auto load_tile = [&](int cc, int buf) {
    const float* rowp = srcB + (size_t)(cc + lc) * LLEN;
    float s = 1.f, h = 0.f;
    if (MODE == 1) { s = sc1[cc + lc]; h = sh1[cc + lc]; }
    for (int f = lq; f < 34; f += 8) {
      int gl0 = l0 - 4 + 4 * f;
      float vv[4];
      if (gl0 >= 0 && gl0 + 3 < LLEN) {
        float4 v4 = *(const float4*)(rowp + gl0);
        vv[0] = v4.x; vv[1] = v4.y; vv[2] = v4.z; vv[3] = v4.w;
      } else {
        #pragma unroll
        for (int j = 0; j < 4; ++j) {
          int gl = gl0 + j;
          vv[j] = (gl >= 0 && gl < LLEN) ? rowp[gl] : 0.f;
        }
      }
      #pragma unroll
      for (int j = 0; j < 4; ++j) {
        float v = vv[j];
        if (MODE == 1) {
          v = v * s + h;                       // fused BN1
          v = fminf(1.f, fmaxf(-1.f, v));      // fused hardtanh
        }
        int ll = 4 * f + j - 3;
        if (ll >= 0 && ll < 130) Bs[buf][ll * KP + lc] = (_Float16)v;
      }
    }
  };

  load_tile(0, 0);

  #pragma unroll
  for (int ccs = 0; ccs < CCH / KC; ++ccs) {
    const int cc  = ccs * KC;
    const int cur = ccs & 1;      // compile-time after unroll -> immediate LDS bases
    __syncthreads();
    if (cc + KC < CCH) load_tile(cc + KC, cur ^ 1);
    // prefetch two chunks ahead (global_prefetch_b8)
    if (cc + 2 * KC < CCH) {
      int po = l0 - 4 + (tid >> 5) * 17;
      if (po < 0) po = 0;
      if (po < LLEN)
        __builtin_prefetch(srcB + (size_t)(cc + 2 * KC + (tid & 31)) * LLEN + po, 0, 1);
    }

    #pragma unroll
    for (int k = 0; k < 3; ++k) {
      AF a[2];
      #pragma unroll
      for (int mf = 0; mf < 2; ++mf) {
        int M = m0 + wm * 32 + mf * 16 + col;
        const v8h* ap = (const v8h*)(wsgn + ((size_t)(k * CCH + M)) * CCH + cc + hi * 8);
        a[mf].h[0] = ap[0];   // K = base .. base+7
        a[mf].h[1] = ap[2];   // K = base+16 .. base+23
      }
      #pragma unroll
      for (int nf = 0; nf < 4; ++nf) {
        AF bf;
        const _Float16* bp = &Bs[cur][(wn * 64 + nf * 16 + col + k) * KP + hi * 8];
        bf.h[0] = *(const v8h*)bp;
        bf.h[1] = *(const v8h*)(bp + 16);
        #pragma unroll
        for (int mf = 0; mf < 2; ++mf) {
          acc[mf][nf] = __builtin_amdgcn_wmma_f32_16x16x32_f16(
              false, a[mf].v, false, bf.v, (short)0, acc[mf][nf], false, false);
        }
      }
    }
  }

  // Store raw conv output + wave-level channel reductions for BN (and SE row sums)
  #pragma unroll
  for (int mf = 0; mf < 2; ++mf) {
    float psum[8], psq[8];
    #pragma unroll
    for (int r = 0; r < 8; ++r) { psum[r] = 0.f; psq[r] = 0.f; }
    #pragma unroll
    for (int nf = 0; nf < 4; ++nf) {
      int n  = l0 + wn * 64 + nf * 16 + col;
      int ch = m0 + wm * 32 + mf * 16 + hi * 8;
      float* dp = dst + ((size_t)b * CCH + ch) * LLEN + n;
      #pragma unroll
      for (int r = 0; r < 8; ++r) {
        float v = acc[mf][nf][r];
        dp[(size_t)r * LLEN] = v;
        psum[r] += v;
        psq[r]  += v * v;
      }
    }
    #pragma unroll
    for (int r = 0; r < 8; ++r) {
      #pragma unroll
      for (int m = 8; m >= 1; m >>= 1) {
        psum[r] += __shfl_xor(psum[r], m, 32);
        psq[r]  += __shfl_xor(psq[r],  m, 32);
      }
    }
    if ((lane & 15) == 0) {
      int chb = m0 + wm * 32 + mf * 16 + hi * 8;
      #pragma unroll
      for (int r = 0; r < 8; ++r) {
        atomicAdd(&csum[chb + r],   psum[r]);
        atomicAdd(&csumsq[chb + r], psq[r]);
        if (MODE == 1) atomicAdd(&rsum[b * CCH + chb + r], psum[r]);
      }
    }
  }
}

// ---------------- BN finalize: scale/shift from batch stats ----------------
__global__ void k_bn1(const float* __restrict__ csum, const float* __restrict__ csumsq,
                      const float* __restrict__ g, const float* __restrict__ be,
                      float* __restrict__ sc, float* __restrict__ sh) {
  int c = threadIdx.x;
  float mean = csum[c] * (1.f / (float)NLTOT);
  float var  = csumsq[c] * (1.f / (float)NLTOT) - mean * mean;
  float s = g[c] * rsqrtf(var + 1e-5f);
  sc[c] = s;
  sh[c] = be[c] - mean * s;
}

// ---------------- BN2 params + SE MLP (one block per batch) ----------------
__global__ void k_se(const float* __restrict__ csum2, const float* __restrict__ csumsq2,
                     const float* __restrict__ g2, const float* __restrict__ b2,
                     const float* __restrict__ rsum,
                     const float* __restrict__ fc1, const float* __restrict__ fc2,
                     float* __restrict__ sc2o, float* __restrict__ sh2o,
                     float* __restrict__ sfac) {
  __shared__ float z[CCH];
  __shared__ float hbuf[64];
  int b = blockIdx.x;
  int c = threadIdx.x;
  float mean = csum2[c] * (1.f / (float)NLTOT);
  float var  = csumsq2[c] * (1.f / (float)NLTOT) - mean * mean;
  float s  = g2[c] * rsqrtf(var + 1e-5f);
  float sh = b2[c] - mean * s;
  if (b == 0) { sc2o[c] = s; sh2o[c] = sh; }
  z[c] = s * (rsum[b * CCH + c] * (1.f / (float)LLEN)) + sh;  // mean over L of bn2(y2)
  __syncthreads();
  if (c < 64) {
    float a = 0.f;
    for (int j = 0; j < CCH; ++j) a += fc1[c * CCH + j] * z[j];
    hbuf[c] = fmaxf(a, 0.f);
  }
  __syncthreads();
  float a = 0.f;
  for (int j = 0; j < 64; ++j) a += fc2[c * 64 + j] * hbuf[j];
  sfac[b * CCH + c] = 1.f / (1.f + expf(-a));
}

// ---------------- fused epilogue: out = hardtanh(bn2(y2)*s + x), in place ----------------
__global__ void k_final(float* __restrict__ out, const float* __restrict__ x,
                        const float* __restrict__ sc2, const float* __restrict__ sh2,
                        const float* __restrict__ sfac) {
  size_t i4 = (size_t)blockIdx.x * blockDim.x + threadIdx.x;  // float4 index
  size_t idx = i4 * 4;
  int c = (int)((idx / LLEN) & (CCH - 1));
  int b = (int)(idx / ((size_t)CCH * LLEN));
  float s = sc2[c], sh = sh2[c], f = sfac[b * CCH + c];
  float4 y  = ((const float4*)out)[i4];
  float4 xi = ((const float4*)x)[i4];
  float4 o;
  o.x = fminf(1.f, fmaxf(-1.f, (y.x * s + sh) * f + xi.x));
  o.y = fminf(1.f, fmaxf(-1.f, (y.y * s + sh) * f + xi.y));
  o.z = fminf(1.f, fmaxf(-1.f, (y.z * s + sh) * f + xi.z));
  o.w = fminf(1.f, fmaxf(-1.f, (y.w * s + sh) * f + xi.w));
  ((float4*)out)[i4] = o;
}

extern "C" void kernel_launch(void* const* d_in, const int* in_sizes, int n_in,
                              void* d_out, int out_size, void* d_ws, size_t ws_size,
                              hipStream_t stream) {
  (void)in_sizes; (void)n_in; (void)out_size; (void)ws_size;
  const float* x   = (const float*)d_in[0];
  const float* w1  = (const float*)d_in[1];
  const float* g1  = (const float*)d_in[2];
  const float* b1  = (const float*)d_in[3];
  const float* w2  = (const float*)d_in[4];
  const float* g2  = (const float*)d_in[5];
  const float* b2  = (const float*)d_in[6];
  const float* fc1 = (const float*)d_in[7];
  const float* fc2 = (const float*)d_in[8];
  float* out = (float*)d_out;

  char* wsb = (char*)d_ws;
  size_t y1bytes = (size_t)BATCH * CCH * LLEN * sizeof(float);   // 128 MB
  float*    y1  = (float*)wsb;
  _Float16* ws1 = (_Float16*)(wsb + y1bytes);
  _Float16* ws2 = ws1 + 3 * CCH * CCH;
  float* stats  = (float*)(wsb + y1bytes + (size_t)2 * 3 * CCH * CCH * sizeof(_Float16));
  float* csum1 = stats;
  float* csq1  = stats + 256;
  float* sc1   = stats + 512;
  float* sh1   = stats + 768;
  float* csum2 = stats + 1024;
  float* csq2  = stats + 1280;
  float* rsum  = stats + 1536;                 // 32*256
  float* sfac  = stats + 1536 + BATCH * CCH;   // 32*256
  float* sc2   = stats + 1536 + 2 * BATCH * CCH;
  float* sh2   = sc2 + 256;
  int statn = 1536 + 2 * BATCH * CCH + 512;    // 18432 floats

  k_zero<<<(statn + 255) / 256, 256, 0, stream>>>(stats, statn);
  k_sign<<<(3 * CCH * CCH + 255) / 256, 256, 0, stream>>>(w1, w2, ws1, ws2);

  dim3 grid(CCH / TM, LLEN / TN, BATCH);  // (2, 32, 32)
  k_conv_gemm<0><<<grid, 256, 0, stream>>>(x,  ws1, y1,  nullptr, nullptr,
                                           csum1, csq1, nullptr);
  k_bn1<<<1, 256, 0, stream>>>(csum1, csq1, g1, b1, sc1, sh1);
  k_conv_gemm<1><<<grid, 256, 0, stream>>>(y1, ws2, out, sc1, sh1,
                                           csum2, csq2, rsum);
  k_se<<<BATCH, 256, 0, stream>>>(csum2, csq2, g2, b2, rsum, fc1, fc2, sc2, sh2, sfac);
  k_final<<<(BATCH * CCH * LLEN / 4) / 256, 256, 0, stream>>>(out, x, sc2, sh2, sfac);
}
